// GNNModule_128849019482
// MI455X (gfx1250) — matrix operand order repaired and checked by reading, not compiled, not run on hardware
//
#include <hip/hip_runtime.h>
#include <hip/hip_bf16.h>

typedef __attribute__((ext_vector_type(16))) __bf16 v16bf;
typedef __attribute__((ext_vector_type(8)))  float  v8f;
typedef __attribute__((ext_vector_type(4)))  float  v4f;

#define NROWS 8192
#define KTOT  8192
#define EPS   1e-5f

// ---------------------------------------------------------------------------
// Kernel 1: prep.  Z = x @ [Wx|Wxr].T + (bx + sum_r bl) ; Y_r = x @ [Wl_r|Wlr_r].T
// Y is stored hi/lo bf16, TRANSPOSED: Yt[r][col(64)][k(8192)] for contiguous
// WMMA B-tile loads in the main kernel.
// grid = (32, 4): blockIdx.y == s selects which 64x64 weight matrix (s=0 -> Z,
// s=1..3 -> Y_{s-1}); each thread owns one row k of x.
// ---------------------------------------------------------------------------
__global__ __launch_bounds__(256) void prep_kernel(
    const float* __restrict__ x,
    const float* __restrict__ Wx,  const float* __restrict__ bx,
    const float* __restrict__ Wxr, const float* __restrict__ bxr,
    const float* __restrict__ Wl,  const float* __restrict__ bl,
    const float* __restrict__ Wlr, const float* __restrict__ blr,
    float* __restrict__ Z, __bf16* __restrict__ Yth, __bf16* __restrict__ Ytl)
{
    __shared__ __align__(16) float Wc[64 * 64];
    const int s = blockIdx.y;

    // Stage the 64x64 combined weight matrix for this s into LDS.
    for (int i = threadIdx.x; i < 64 * 64; i += 256) {
        int c = i >> 6, f = i & 63;
        float v;
        if (s == 0) {
            v = (c < 32) ? Wx[c * 64 + f] : Wxr[(c - 32) * 64 + f];
        } else {
            int r = s - 1;
            v = (c < 32) ? Wl[(r * 32 + c) * 64 + f]
                         : Wlr[(r * 32 + (c - 32)) * 64 + f];
        }
        Wc[i] = v;
    }
    __syncthreads();

    const int k = blockIdx.x * 256 + threadIdx.x;   // row of x

    v4f xv[16];
#pragma unroll
    for (int i = 0; i < 16; ++i)
        xv[i] = *(const v4f*)(x + (size_t)k * 64 + 4 * i);

    for (int c = 0; c < 64; ++c) {
        float acc = 0.f;
        const float* wr = &Wc[c * 64];
#pragma unroll
        for (int f4 = 0; f4 < 16; ++f4) {
            v4f wv = *(const v4f*)(wr + 4 * f4);
            acc = fmaf(xv[f4][0], wv[0], acc);
            acc = fmaf(xv[f4][1], wv[1], acc);
            acc = fmaf(xv[f4][2], wv[2], acc);
            acc = fmaf(xv[f4][3], wv[3], acc);
        }
        if (s == 0) {
            float bias = (c < 32)
                ? (bx[c]       + bl[c]        + bl[32 + c]        + bl[64 + c])
                : (bxr[c - 32] + blr[c - 32]  + blr[32 + c - 32]  + blr[64 + c - 32]);
            Z[(size_t)k * 64 + c] = acc + bias;
        } else {
            int r = s - 1;
            __bf16 h = (__bf16)acc;                 // hi part (RNE)
            float  rem = acc - (float)h;            // residual
            size_t off = ((size_t)(r * 64 + c)) * KTOT + k;
            Yth[off] = h;
            Ytl[off] = (__bf16)rem;                 // lo part
        }
    }
}

// ---------------------------------------------------------------------------
// Kernel 2: the heavy GEMM.  cat = Z + sum_r w_r @ Y_r, relu on cols 32..63.
// One wave per 16-row x 64-col output strip; K loop in steps of 32 with
// v_wmma_f32_16x16x32_bf16, split-precision (AhBh + AhBl + AlBh).
// 64 blocks x 256 threads (8 waves) = 512 strips = 8192 rows.
// ---------------------------------------------------------------------------
__global__ __launch_bounds__(256) void gemm_kernel(
    const float* __restrict__ w0, const float* __restrict__ w1,
    const float* __restrict__ w2,
    const float* __restrict__ Z,
    const __bf16* __restrict__ Yth, const __bf16* __restrict__ Ytl,
    float* __restrict__ catb)
{
    const int lane = threadIdx.x & 31;
    const int wave = threadIdx.x >> 5;
    const int tile = blockIdx.x * 8 + wave;
    const int row0 = tile * 16;
    const int m    = lane & 15;          // A row / C col index
    const int g    = lane >> 4;          // lane half

    // Seed accumulators with Z (C/D layout: m = v + g*8, n = t*16 + m)
    v8f acc[4];
#pragma unroll
    for (int t = 0; t < 4; ++t) {
#pragma unroll
        for (int v = 0; v < 8; ++v)
            acc[t][v] = Z[(size_t)(row0 + v + g * 8) * 64 + t * 16 + m];
    }

#pragma unroll 1
    for (int r = 0; r < 3; ++r) {
        const float*  wmat = (r == 0) ? w0 : ((r == 1) ? w1 : w2);
        const float*  wr   = wmat + (size_t)(row0 + m) * KTOT;
        const __bf16* yh   = Yth + (size_t)r * 64 * KTOT;
        const __bf16* yl   = Ytl + (size_t)r * 64 * KTOT;

        for (int kb = 0; kb < KTOT; kb += 32) {
            // ---- A tile: 16x32 f32 from w, split into bf16 hi/lo in-register.
            // Lane layout: K = g*16 + j (contiguous, matches B)
            __builtin_prefetch(wr + kb + 256, 0, 0);      // stream-ahead ~1KB
            v8f a0 = __builtin_nontemporal_load((const v8f*)(wr + kb + g * 16));
            v8f a1 = __builtin_nontemporal_load((const v8f*)(wr + kb + g * 16 + 8));
            v16bf ah, al;
#pragma unroll
            for (int i = 0; i < 8; ++i) {
                float f0 = a0[i], f1 = a1[i];
                __bf16 h0 = (__bf16)f0, h1 = (__bf16)f1;
                ah[i]     = h0;
                ah[i + 8] = h1;
                al[i]     = (__bf16)(f0 - (float)h0);
                al[i + 8] = (__bf16)(f1 - (float)h1);
            }

            // ---- B tiles: 32x16 bf16, lane holds col n = t*16+m, K = kb + g*16 + j
            v16bf bh[4], bl2[4];
#pragma unroll
            for (int t = 0; t < 4; ++t) {
                size_t off = (size_t)(t * 16 + m) * KTOT + kb + g * 16;
                bh[t]  = *(const v16bf*)(yh + off);
                bl2[t] = *(const v16bf*)(yl + off);
            }

            // ---- 12 WMMAs, interleaved across the 4 independent accumulators
#pragma unroll
            for (int t = 0; t < 4; ++t)
                acc[t] = __builtin_amdgcn_wmma_f32_16x16x32_bf16(
                    false, ah, false, bh[t], (short)0, acc[t], false, false);
#pragma unroll
            for (int t = 0; t < 4; ++t)
                acc[t] = __builtin_amdgcn_wmma_f32_16x16x32_bf16(
                    false, ah, false, bl2[t], (short)0, acc[t], false, false);
#pragma unroll
            for (int t = 0; t < 4; ++t)
                acc[t] = __builtin_amdgcn_wmma_f32_16x16x32_bf16(
                    false, al, false, bh[t], (short)0, acc[t], false, false);
        }
    }

    // Epilogue: relu on channels 32..63 (t = 2,3), write cat.
#pragma unroll
    for (int t = 0; t < 4; ++t) {
#pragma unroll
        for (int v = 0; v < 8; ++v) {
            float val = acc[t][v];
            if (t >= 2) val = fmaxf(val, 0.f);
            catb[(size_t)(row0 + v + g * 8) * 64 + t * 16 + m] = val;
        }
    }
}

// ---------------------------------------------------------------------------
// Kernel 3: per-channel mean / inv-std over N=8192 (deterministic tree reduce)
// ---------------------------------------------------------------------------
__global__ __launch_bounds__(256) void stats_kernel(
    const float* __restrict__ catb, float* __restrict__ mean,
    float* __restrict__ istd)
{
    __shared__ float ssum[256];
    __shared__ float ssq[256];
    const int c = blockIdx.x, tid = threadIdx.x;
    float s = 0.f, q = 0.f;
    for (int k = tid; k < NROWS; k += 256) {
        float v = catb[(size_t)k * 64 + c];
        s += v;
        q = fmaf(v, v, q);
    }
    ssum[tid] = s; ssq[tid] = q;
    __syncthreads();
    for (int st = 128; st > 0; st >>= 1) {
        if (tid < st) { ssum[tid] += ssum[tid + st]; ssq[tid] += ssq[tid + st]; }
        __syncthreads();
    }
    if (tid == 0) {
        float mu  = ssum[0] * (1.f / NROWS);
        float var = ssq[0] * (1.f / NROWS) - mu * mu;   // biased, matches BN
        mean[c] = mu;
        istd[c] = rsqrtf(var + EPS);
    }
}

// ---------------------------------------------------------------------------
// Kernel 4: normalize (coalesced row-major)
// ---------------------------------------------------------------------------
__global__ __launch_bounds__(256) void bn_kernel(
    const float* __restrict__ catb, const float* __restrict__ mean,
    const float* __restrict__ istd, const float* __restrict__ gamma,
    const float* __restrict__ beta, float* __restrict__ out)
{
    int i = blockIdx.x * 256 + threadIdx.x;
    int c = i & 63;
    out[i] = fmaf(gamma[c] * istd[c], catb[i] - mean[c], beta[c]);
}

// ---------------------------------------------------------------------------
extern "C" void kernel_launch(void* const* d_in, const int* in_sizes, int n_in,
                              void* d_out, int out_size, void* d_ws, size_t ws_size,
                              hipStream_t stream)
{
    const float* x     = (const float*)d_in[0];
    const float* w0    = (const float*)d_in[1];
    const float* w1    = (const float*)d_in[2];
    const float* w2    = (const float*)d_in[3];
    const float* Wx    = (const float*)d_in[4];
    const float* bx    = (const float*)d_in[5];
    const float* Wxr   = (const float*)d_in[6];
    const float* bxr   = (const float*)d_in[7];
    const float* Wl    = (const float*)d_in[8];
    const float* bl    = (const float*)d_in[9];
    const float* Wlr   = (const float*)d_in[10];
    const float* blr   = (const float*)d_in[11];
    const float* gamma = (const float*)d_in[12];
    const float* beta  = (const float*)d_in[13];
    float* out = (float*)d_out;

    // Workspace layout (~10.5 MB total)
    char*   base = (char*)d_ws;
    float*  Z    = (float*)(base);                          // 2 MB
    __bf16* Yth  = (__bf16*)(base + ((size_t)2  << 20));    // 3 MB
    __bf16* Ytl  = (__bf16*)(base + ((size_t)5  << 20));    // 3 MB
    float*  catb = (float*)(base + ((size_t)8  << 20));     // 2 MB
    float*  mean = (float*)(base + ((size_t)10 << 20));     // 64 f
    float*  istd = mean + 64;                               // 64 f

    prep_kernel<<<dim3(32, 4), 256, 0, stream>>>(x, Wx, bx, Wxr, bxr,
                                                 Wl, bl, Wlr, blr, Z, Yth, Ytl);
    gemm_kernel<<<64, 256, 0, stream>>>(w0, w1, w2, Z, Yth, Ytl, catb);
    stats_kernel<<<64, 256, 0, stream>>>(catb, mean, istd);
    bn_kernel<<<(NROWS * 64) / 256, 256, 0, stream>>>(catb, mean, istd,
                                                      gamma, beta, out);
}
